// NNAttr_78408922956189
// MI455X (gfx1250) — matrix operand order repaired and checked by reading, not compiled, not run on hardware
//
#include <hip/hip_runtime.h>

// ---------------------------------------------------------------------------
// CDNA5 (gfx1250) implementation of the NNConv+GRU+mixture-MLP reference.
// Heavy contractions use v_wmma_f32_16x16x32_bf16 (wave32 WMMA).
// GEMM tiles are staged with GLOBAL_LOAD_ASYNC_TO_LDS_B128 (ASYNCcnt) into
// double-buffered LDS so the DMA overlaps the WMMA stream.
// Workspace budget: ~60 MB (bump-allocated from d_ws).
// ---------------------------------------------------------------------------

typedef unsigned short ushort_t;                                  // raw bf16 bits
typedef __attribute__((ext_vector_type(16))) __bf16        v16bf;
typedef __attribute__((ext_vector_type(8)))  float         v8f;
typedef __attribute__((ext_vector_type(4)))  unsigned int  uint4v;

union FragU { v16bf v; uint4v q[2]; };

__device__ __forceinline__ ushort_t f2bf(float f) {
  union { float f; unsigned int u; } c; c.f = f;
  unsigned int u = c.u;
  unsigned int r = u + 0x7FFFu + ((u >> 16) & 1u);   // round-to-nearest-even
  return (ushort_t)(r >> 16);
}

__device__ __forceinline__ float sigmoidf_(float x) {
  return 1.0f / (1.0f + __expf(-x));
}

// Async global->LDS 16B copy (CDNA5 TDM-adjacent path, ASYNCcnt-tracked).
// LDS operand = byte offset = low 32 bits of the flat shared pointer.
__device__ __forceinline__ void async_copy_b128(const void* gaddr, void* lds) {
  const unsigned loff = (unsigned)(unsigned long long)lds;
  asm volatile("global_load_async_to_lds_b128 %0, %1, off"
               :: "v"(loff), "v"(gaddr) : "memory");
}
__device__ __forceinline__ void wait_async0() {
  asm volatile("s_wait_asynccnt 0x0" ::: "memory");
}

// A-fragment (16x32 bf16, M rows x K contraction), per ISA 7.12.2:
// lanes 0-15: row=lane,  K = {0..7, 16..23}; lanes 16-31: row=lane-16, K = {8..15, 24..31}
__device__ __forceinline__ v16bf load_frag_a(const ushort_t* base, int ldk, int dbase) {
  const int lane = threadIdx.x & 31;
  const int row  = lane & 15;
  const int koff = (lane < 16) ? 0 : 8;
  const ushort_t* p = base + (size_t)row * ldk + dbase + koff;
  FragU f;
  f.q[0] = *(const uint4v*)(p);
  f.q[1] = *(const uint4v*)(p + 16);
  return f.v;
}

// B-fragment (32x16 bf16, K contraction x N cols) from W-style storage [N][K]:
// lanes 0-15: col=lane, K = {0..7, 16..23}; lanes 16-31: col=lane-16, K = {8..15, 24..31}
__device__ __forceinline__ v16bf load_frag_b(const ushort_t* wbase, int ldk, int nbase, int dbase) {
  const int lane = threadIdx.x & 31;
  const int col  = lane & 15;
  const int koff = (lane < 16) ? 0 : 8;
  const ushort_t* p = wbase + (size_t)(nbase + col) * ldk + dbase + koff;
  FragU f;
  f.q[0] = *(const uint4v*)(p);
  f.q[1] = *(const uint4v*)(p + 16);
  return f.v;
}

// ---------------------------------------------------------------------------
// Generic C = A @ W^T (+bias, optional ReLU, optional bf16 mirror), batched.
// A: [M,K] bf16 row-major (lda), W: [N,K] bf16 row-major (ldw), C: [M,N] f32.
// REQUIRES: M%64==0, N%64==0, K%32==0, lda%8==0, ldw%8==0 (pre-padded inputs).
// Block = 256 threads (8 waves); block tile 64x64; wave tile 16x32 (2 WMMAs).
// Double-buffered async global->LDS staging.
// ---------------------------------------------------------------------------
#define LDSS 40   // LDS row stride in ushorts (16B-aligned, conflict-skewed)

__global__ __launch_bounds__(256) void k_gemm_bf16(
    const ushort_t* __restrict__ A, long long sAb, int lda,
    const ushort_t* __restrict__ W, long long sWb, int ldw,
    const float* __restrict__ bias, long long sBiasb,
    float* __restrict__ C, long long sCb,
    ushort_t* __restrict__ Cbf, long long sCbfb,
    int M, int N, int K, int relu)
{
  __shared__ ushort_t As[2][64 * LDSS];
  __shared__ ushort_t Bs[2][64 * LDSS];
  const int t    = threadIdx.x;
  const int lane = t & 31;
  const int w    = t >> 5;
  const int bz   = blockIdx.z;
  const int m0   = blockIdx.y * 64;
  const int n0   = blockIdx.x * 64;
  const ushort_t* Ab = A + (size_t)bz * sAb;
  const ushort_t* Wb = W + (size_t)bz * sWb;

  const int mt = w & 3;          // 0..3  -> m sub-tile
  const int nb = (w >> 2) * 32;  // 0/32  -> n sub-pair

  // per-thread staging coordinates (8 bf16 = 16B per thread per tile)
  const int srow = t >> 2;          // 0..63
  const int scol = (t & 3) * 8;     // 0,8,16,24
  const ushort_t* ap = Ab + (size_t)(m0 + srow) * lda + scol;
  const ushort_t* wp = Wb + (size_t)(n0 + srow) * ldw + scol;
  ushort_t* asl[2] = { &As[0][srow * LDSS + scol], &As[1][srow * LDSS + scol] };
  ushort_t* bsl[2] = { &Bs[0][srow * LDSS + scol], &Bs[1][srow * LDSS + scol] };

  v8f acc0 = {0,0,0,0,0,0,0,0};
  v8f acc1 = {0,0,0,0,0,0,0,0};

  // prime buffer 0
  async_copy_b128(ap, asl[0]);
  async_copy_b128(wp, bsl[0]);

  int cur = 0;
  for (int kk = 0; kk < K; kk += 32) {
    wait_async0();          // own tile fraction landed in LDS
    __syncthreads();        // whole tile landed
    if (kk + 32 < K) {      // kick off next tile into the other buffer
      async_copy_b128(ap + kk + 32, asl[cur ^ 1]);
      async_copy_b128(wp + kk + 32, bsl[cur ^ 1]);
    }
    v16bf a  = load_frag_a(&As[cur][mt * 16 * LDSS], LDSS, 0);
    v16bf b0 = load_frag_b(&Bs[cur][0], LDSS, nb,      0);
    v16bf b1 = load_frag_b(&Bs[cur][0], LDSS, nb + 16, 0);
    acc0 = __builtin_amdgcn_wmma_f32_16x16x32_bf16(false, a, false, b0, (short)0, acc0, false, false);
    acc1 = __builtin_amdgcn_wmma_f32_16x16x32_bf16(false, a, false, b1, (short)0, acc1, false, false);
    __syncthreads();        // done reading 'cur' before it is refilled next iter
    cur ^= 1;
  }

  const float* bp  = bias ? bias + (size_t)bz * sBiasb : nullptr;
  float*    cp  = C   + (size_t)bz * sCb;
  ushort_t* cbp = Cbf ? Cbf + (size_t)bz * sCbfb : nullptr;
  const int rbase = (lane < 16) ? 0 : 8;
  const int coln  = lane & 15;
#pragma unroll
  for (int r = 0; r < 8; ++r) {
    const int gm = m0 + mt * 16 + r + rbase;
    {
      const int gn = n0 + nb + coln;
      float v = acc0[r] + (bp ? bp[gn] : 0.0f);
      if (relu) v = v > 0.0f ? v : 0.0f;
      cp[(size_t)gm * N + gn] = v;
      if (cbp) cbp[(size_t)gm * N + gn] = f2bf(v);
    }
    {
      const int gn = n0 + nb + 16 + coln;
      float v = acc1[r] + (bp ? bp[gn] : 0.0f);
      if (relu) v = v > 0.0f ? v : 0.0f;
      cp[(size_t)gm * N + gn] = v;
      if (cbp) cbp[(size_t)gm * N + gn] = f2bf(v);
    }
  }
}

// ---------------------------------------------------------------------------
// Fused NNConv message kernel.
//   msg[e,f] = sum_k hidden[e,k] * (Xsrc[e,:] @ W2t[f,k,:])  +  biasm[src[e],f]
//   agg[dst[e],f] += msg[e,f]   (atomic; mean-divide done later)
// W2t layout: [f][k][d] bf16  (W2t[(f*128+k)*256+d] = nn_w2[(d*256+f)*128+k])
// One block per 16-edge tile; wave w owns features f = w, w+8, ...
// Per wave: 32 f * 8 k-tiles * 8 d-steps = 2048 WMMAs; A frags preloaded once.
// W2t panels are L2-resident (16 MB << 192 MB); prefetch the next panel.
// ---------------------------------------------------------------------------
__global__ __launch_bounds__(256) void k_conv_msg(
    const ushort_t* __restrict__ xcb, const float* __restrict__ hidden,
    const int* __restrict__ src, const int* __restrict__ dst,
    const ushort_t* __restrict__ w2t, const float* __restrict__ biasm,
    float* __restrict__ agg)
{
  __shared__ ushort_t Xs[16 * 256];   // gathered x[src] tile, bf16
  __shared__ float    Hs[16 * 128];   // hidden tile, f32
  __shared__ int sIdx[16], dIdx[16];
  const int t = threadIdx.x, lane = t & 31, w = t >> 5;
  const int e0 = blockIdx.x * 16;

  if (t < 16) { sIdx[t] = src[e0 + t]; dIdx[t] = dst[e0 + t]; }
  {
    const int row = t >> 4, seg = t & 15;
    const int s = src[e0 + row];
    const uint4v* gp = (const uint4v*)(xcb + (size_t)s * 256 + seg * 16);
    uint4v* lp = (uint4v*)(&Xs[row * 256 + seg * 16]);
    lp[0] = gp[0]; lp[1] = gp[1];
  }
  {
    const int pos = t * 8, row = pos >> 7, col = pos & 127;
    const float* gp = hidden + (size_t)(e0 + row) * 128 + col;
#pragma unroll
    for (int i = 0; i < 8; ++i) Hs[pos + i] = gp[i];
  }
  __syncthreads();

  v16bf a[8];
#pragma unroll
  for (int s = 0; s < 8; ++s) a[s] = load_frag_a(Xs, 256, s * 32);

  const int rbase = (lane < 16) ? 0 : 8;
  const int kl    = lane & 15;

  for (int f = w; f < 256; f += 8) {
    const ushort_t* wf = w2t + (size_t)f * 128 * 256;
    __builtin_prefetch(wf + (size_t)8 * 128 * 256, 0, 1);   // next panel for this wave
    float msum[8];
#pragma unroll
    for (int r = 0; r < 8; ++r) msum[r] = 0.0f;

    for (int kt = 0; kt < 8; ++kt) {
      v8f acc = {0,0,0,0,0,0,0,0};
#pragma unroll
      for (int s = 0; s < 8; ++s) {
        v16bf b = load_frag_b(wf, 256, kt * 16, s * 32);  // L2-resident nn_w2
        acc = __builtin_amdgcn_wmma_f32_16x16x32_bf16(false, a[s], false, b, (short)0, acc, false, false);
      }
#pragma unroll
      for (int r = 0; r < 8; ++r) {
        const int e = r + rbase;
        const int k = kt * 16 + kl;
        msum[r] += acc[r] * Hs[e * 128 + k];
      }
    }
#pragma unroll
    for (int r = 0; r < 8; ++r) {
      float v = msum[r];
      v += __shfl_xor(v, 1, 32);
      v += __shfl_xor(v, 2, 32);
      v += __shfl_xor(v, 4, 32);
      v += __shfl_xor(v, 8, 32);
      if (kl == 0) {
        const int e = r + rbase;
        const float total = v + biasm[(size_t)sIdx[e] * 256 + f];
        atomicAdd(&agg[(size_t)dIdx[e] * 256 + f], total);
      }
    }
  }
}

// ------------------------------ small kernels ------------------------------

__global__ void k_convert_bf16(const float* __restrict__ in, ushort_t* __restrict__ out, int n) {
  int i = blockIdx.x * 256 + threadIdx.x;
  if (i < n) out[i] = f2bf(in[i]);
}

// row-padded convert: out[r*outcols+c] = c < incols ? bf16(in[r*incols+c]) : 0
__global__ void k_convert_pad(const float* __restrict__ in, ushort_t* __restrict__ out,
                              int rows, int incols, int outcols) {
  int i = blockIdx.x * 256 + threadIdx.x;
  if (i >= rows * outcols) return;
  const int r = i / outcols, c = i - r * outcols;
  out[i] = (c < incols) ? f2bf(in[(size_t)r * incols + c]) : (ushort_t)0;
}

__global__ void k_zero(float* p, int n) {
  int i = blockIdx.x * 256 + threadIdx.x;
  if (i < n) p[i] = 0.0f;
}

// W2t[(f*128+k)*256+d] = nn_w2[(d*256+f)*128+k], converted to bf16
__global__ void k_w2_transpose(const float* __restrict__ nn_w2, ushort_t* __restrict__ w2t) {
  int i = blockIdx.x * 256 + threadIdx.x;
  if (i >= 256 * 128 * 256) return;
  const int d = i & 255, k = (i >> 8) & 127, f = i >> 15;
  w2t[i] = f2bf(nn_w2[((size_t)d * 256 + f) * 128 + k]);
}

// b2t[f*256+d] = nn_b2[d*256+f]  (W-form for the nn_b2 GEMM)
__global__ void k_b2_transpose(const float* __restrict__ nn_b2, ushort_t* __restrict__ b2t) {
  int i = blockIdx.x * 256 + threadIdx.x;
  if (i >= 65536) return;
  const int d = i & 255, f = i >> 8;
  b2t[i] = f2bf(nn_b2[(size_t)d * 256 + f]);
}

__global__ void k_bn_stats(const float* __restrict__ y, float* __restrict__ mean, float* __restrict__ var) {
  __shared__ float rs[256], rs2[256];
  const int c = blockIdx.x;
  float s = 0.0f, s2 = 0.0f;
  for (int r = threadIdx.x; r < 1024; r += 256) {
    const float v = y[(size_t)r * 256 + c];
    s += v; s2 += v * v;
  }
  rs[threadIdx.x] = s; rs2[threadIdx.x] = s2;
  __syncthreads();
  for (int o = 128; o > 0; o >>= 1) {
    if (threadIdx.x < o) { rs[threadIdx.x] += rs[threadIdx.x + o]; rs2[threadIdx.x] += rs2[threadIdx.x + o]; }
    __syncthreads();
  }
  if (threadIdx.x == 0) {
    const float m = rs[0] * (1.0f / 1024.0f);
    mean[c] = m;
    var[c]  = rs2[0] * (1.0f / 1024.0f) - m * m;
  }
}

__global__ void k_bn_apply(const float* __restrict__ y, const float* __restrict__ mean,
                           const float* __restrict__ var, const float* __restrict__ g,
                           const float* __restrict__ b, float* __restrict__ h, ushort_t* __restrict__ hb) {
  int i = blockIdx.x * 256 + threadIdx.x;
  if (i >= 1024 * 256) return;
  const int c = i & 255;
  const float v = (y[i] - mean[c]) * rsqrtf(var[c] + 1e-5f) * g[c] + b[c];
  h[i] = v; hb[i] = f2bf(v);
}

__global__ void k_edge_mlp(const float* __restrict__ ea, const float* __restrict__ w1,
                           const float* __restrict__ b1, float* __restrict__ hidden) {
  int i = blockIdx.x * 256 + threadIdx.x;
  if (i >= 2048 * 128) return;
  const int e = i >> 7, k = i & 127;
  const float* a  = ea + (size_t)e * 4;
  const float* wv = w1 + (size_t)k * 4;
  const float s = b1[k] + a[0]*wv[0] + a[1]*wv[1] + a[2]*wv[2] + a[3]*wv[3];
  hidden[i] = sigmoidf_(s);
}

__global__ void k_count(const int* __restrict__ dst, float* __restrict__ cnt) {
  int e = blockIdx.x * 256 + threadIdx.x;
  if (e < 2048) atomicAdd(&cnt[dst[e]], 1.0f);
}

// m = relu(agg/max(cnt,1) + (xc@root_w.T + conv_b))   -> bf16 (GRU input)
__global__ void k_combine(const float* __restrict__ agg, const float* __restrict__ root,
                          const float* __restrict__ cnt, ushort_t* __restrict__ mb) {
  int i = blockIdx.x * 256 + threadIdx.x;
  if (i >= 1024 * 256) return;
  const int n = i >> 8;
  const float dnm = cnt[n] > 1.0f ? cnt[n] : 1.0f;
  float v = agg[i] / dnm + root[i];
  v = v > 0.0f ? v : 0.0f;
  mb[i] = f2bf(v);
}

__global__ void k_gru(const float* __restrict__ gi, const float* __restrict__ gh,
                      float* __restrict__ h, ushort_t* __restrict__ hb) {
  int i = blockIdx.x * 256 + threadIdx.x;
  if (i >= 1024 * 256) return;
  const int n = i >> 8, j = i & 255;
  const float* gin = gi + (size_t)n * 768;
  const float* ghn = gh + (size_t)n * 768;
  const float r  = sigmoidf_(gin[j]       + ghn[j]);
  const float z  = sigmoidf_(gin[256 + j] + ghn[256 + j]);
  const float nn = tanhf(gin[512 + j] + r * ghn[512 + j]);
  const float hv = (1.0f - z) * nn + z * h[i];
  h[i] = hv; hb[i] = f2bf(hv);
}

__global__ void k_mix(const float* __restrict__ h2, const float* __restrict__ w3,
                      const float* __restrict__ b3, float* __restrict__ mix) {
  int i = blockIdx.x * 256 + threadIdx.x;
  if (i >= 10 * 1024) return;
  const int m = i >> 10, n = i & 1023;
  const float* hp = h2 + ((size_t)m * 1024 + n) * 256;
  const float* wp = w3 + (size_t)m * 256;
  float s = b3[m];
  for (int k = 0; k < 256; ++k) s += hp[k] * wp[k];
  mix[i] = s;
}

__global__ void k_musigma(const float* __restrict__ mix, float* __restrict__ out) {
  int n = blockIdx.x * 256 + threadIdx.x;
  if (n >= 1024) return;
  float s = 0.0f;
  for (int m = 0; m < 10; ++m) s += mix[m * 1024 + n];
  const float mu = s * 0.1f;
  float v = 0.0f;
  for (int m = 0; m < 10; ++m) { const float d = mix[m * 1024 + n] - mu; v += d * d; }
  v *= (1.0f / 9.0f);                        // unbiased (ddof=1)
  out[n] = mu;
  out[1024 + n] = sqrtf(v + 1e-5f);
}

// ---------------------------------------------------------------------------

extern "C" void kernel_launch(void* const* d_in, const int* in_sizes, int n_in,
                              void* d_out, int out_size, void* d_ws, size_t ws_size,
                              hipStream_t stream) {
  const float* x        = (const float*)d_in[0];
  const float* ea       = (const float*)d_in[1];
  const int*   ei       = (const int*)  d_in[2];
  const float* lin_w    = (const float*)d_in[3];
  const float* lin_b    = (const float*)d_in[4];
  const float* bn_g     = (const float*)d_in[5];
  const float* bn_b     = (const float*)d_in[6];
  const float* nn_w1    = (const float*)d_in[7];
  const float* nn_b1    = (const float*)d_in[8];
  const float* nn_w2    = (const float*)d_in[9];
  const float* nn_b2    = (const float*)d_in[10];
  const float* root_w   = (const float*)d_in[11];
  const float* conv_b   = (const float*)d_in[12];
  const float* gru_wih  = (const float*)d_in[13];
  const float* gru_whh  = (const float*)d_in[14];
  const float* gru_bih  = (const float*)d_in[15];
  const float* gru_bhh  = (const float*)d_in[16];
  const float* mlp_w1   = (const float*)d_in[17];
  const float* mlp_b1   = (const float*)d_in[18];
  const float* mlp_w2   = (const float*)d_in[19];
  const float* mlp_b2   = (const float*)d_in[20];
  const float* mlp_w3   = (const float*)d_in[21];
  const float* mlp_b3   = (const float*)d_in[22];
  const int* src = ei;
  const int* dst = ei + 2048;
  float* out = (float*)d_out;

  char* wsb = (char*)d_ws;
  size_t off = 0;
  auto alloc = [&](size_t bytes) -> void* {
    void* p = wsb + off;
    off += (bytes + 255) & ~(size_t)255;
    return p;
  };

  float*    y      = (float*)   alloc(1024 * 256 * 4);
  float*    h      = (float*)   alloc(1024 * 256 * 4);
  ushort_t* hb     = (ushort_t*)alloc(1024 * 256 * 2);
  ushort_t* mb     = (ushort_t*)alloc(1024 * 256 * 2);
  float*    rooto  = (float*)   alloc(1024 * 256 * 4);
  float*    biasm  = (float*)   alloc(1024 * 256 * 4);
  float*    agg    = (float*)   alloc(1024 * 256 * 4);
  float*    gi     = (float*)   alloc(1024 * 768 * 4);
  float*    gh     = (float*)   alloc(1024 * 768 * 4);
  float*    hidden = (float*)   alloc(2048 * 128 * 4);
  float*    cnt    = (float*)   alloc(1024 * 4);
  float*    meanb  = (float*)   alloc(256 * 4);
  float*    varb   = (float*)   alloc(256 * 4);
  float*    mixb   = (float*)   alloc(10 * 1024 * 4);
  ushort_t* xb     = (ushort_t*)alloc(1024 * 64 * 2);              // x padded K=38->64
  ushort_t* linwb  = (ushort_t*)alloc(256 * 64 * 2);               // lin_w padded
  ushort_t* rootwb = (ushort_t*)alloc(256 * 256 * 2);
  ushort_t* wihb   = (ushort_t*)alloc(768 * 256 * 2);
  ushort_t* whhb   = (ushort_t*)alloc(768 * 256 * 2);
  ushort_t* w1b    = (ushort_t*)alloc(10 * 256 * 256 * 2);
  ushort_t* w2b    = (ushort_t*)alloc(10 * 256 * 256 * 2);
  ushort_t* b2tb   = (ushort_t*)alloc(256 * 256 * 2);
  ushort_t* w2t    = (ushort_t*)alloc((size_t)256 * 128 * 256 * 2);  // 16 MB, L2-resident
  float*    h1     = (float*)   alloc((size_t)10 * 1024 * 256 * 4);
  ushort_t* h1b    = (ushort_t*)alloc((size_t)10 * 1024 * 256 * 2);
  float*    h2     = (float*)   alloc((size_t)10 * 1024 * 256 * 4);

  auto blocks = [](int n) { return (n + 255) / 256; };

  // ---- one-time (per call) conversions (pad K=38 operands to K=64) ----
  k_convert_pad <<<blocks(1024*64),   256, 0, stream>>>(x,     xb,    1024, 38, 64);
  k_convert_pad <<<blocks(256*64),    256, 0, stream>>>(lin_w, linwb, 256,  38, 64);
  k_convert_bf16<<<blocks(256*256),   256, 0, stream>>>(root_w,  rootwb, 256*256);
  k_convert_bf16<<<blocks(768*256),   256, 0, stream>>>(gru_wih, wihb,   768*256);
  k_convert_bf16<<<blocks(768*256),   256, 0, stream>>>(gru_whh, whhb,   768*256);
  k_convert_bf16<<<blocks(10*256*256),256, 0, stream>>>(mlp_w1,  w1b,    10*256*256);
  k_convert_bf16<<<blocks(10*256*256),256, 0, stream>>>(mlp_w2,  w2b,    10*256*256);
  k_w2_transpose<<<blocks(256*128*256), 256, 0, stream>>>(nn_w2, w2t);
  k_b2_transpose<<<blocks(65536),       256, 0, stream>>>(nn_b2, b2tb);

  // ---- input embed: y = x @ lin_w.T + lin_b  (padded K=64) ----
  k_gemm_bf16<<<dim3(4, 16, 1), 256, 0, stream>>>(
      xb, 0, 64, linwb, 0, 64, lin_b, 0, y, 0, nullptr, 0, 1024, 256, 64, 0);

  // ---- batchnorm (training-mode batch stats) -> h = xh (f32 + bf16) ----
  k_bn_stats<<<256, 256, 0, stream>>>(y, meanb, varb);
  k_bn_apply<<<blocks(1024*256), 256, 0, stream>>>(y, meanb, varb, bn_g, bn_b, h, hb);

  // ---- edge MLP: hidden = sigmoid(edge_attr @ nn_w1.T + nn_b1) ----
  k_edge_mlp<<<blocks(2048*128), 256, 0, stream>>>(ea, nn_w1, nn_b1, hidden);

  // ---- in-degree counts ----
  k_zero <<<blocks(1024), 256, 0, stream>>>(cnt, 1024);
  k_count<<<blocks(2048), 256, 0, stream>>>(dst, cnt);

  // ---- 3 rounds of NNConv + GRU (xc == h throughout) ----
  for (int it = 0; it < 3; ++it) {
    k_zero<<<blocks(1024*256), 256, 0, stream>>>(agg, 1024*256);
    // root term with conv bias folded in: rooto = xc @ root_w.T + conv_b
    k_gemm_bf16<<<dim3(4, 16, 1), 256, 0, stream>>>(
        hb, 0, 256, rootwb, 0, 256, conv_b, 0, rooto, 0, nullptr, 0, 1024, 256, 256, 0);
    // nn_b2 contribution: biasm = xc @ reshape(nn_b2)  (gathered per-edge in conv)
    k_gemm_bf16<<<dim3(4, 16, 1), 256, 0, stream>>>(
        hb, 0, 256, b2tb, 0, 256, nullptr, 0, biasm, 0, nullptr, 0, 1024, 256, 256, 0);
    // fused edge-conditioned messages + scatter-add
    k_conv_msg<<<128, 256, 0, stream>>>(hb, hidden, src, dst, w2t, biasm, agg);
    // m = relu(agg/denom + rooto)  (bf16 for GRU GEMM)
    k_combine<<<blocks(1024*256), 256, 0, stream>>>(agg, rooto, cnt, mb);
    // GRU gates
    k_gemm_bf16<<<dim3(12, 16, 1), 256, 0, stream>>>(
        mb, 0, 256, wihb, 0, 256, gru_bih, 0, gi, 0, nullptr, 0, 1024, 768, 256, 0);
    k_gemm_bf16<<<dim3(12, 16, 1), 256, 0, stream>>>(
        hb, 0, 256, whhb, 0, 256, gru_bhh, 0, gh, 0, nullptr, 0, 1024, 768, 256, 0);
    k_gru<<<blocks(1024*256), 256, 0, stream>>>(gi, gh, h, hb);
  }

  // ---- mixture MLP head (batched over 10 mixtures via grid.z) ----
  k_gemm_bf16<<<dim3(4, 16, 10), 256, 0, stream>>>(
      hb, 0, 256, w1b, 256*256, 256, mlp_b1, 256,
      h1, 1024*256, h1b, 1024*256, 1024, 256, 256, 1);
  k_gemm_bf16<<<dim3(4, 16, 10), 256, 0, stream>>>(
      h1b, 1024*256, 256, w2b, 256*256, 256, mlp_b2, 256,
      h2, 1024*256, nullptr, 0, 1024, 256, 256, 1);
  k_mix<<<blocks(10*1024), 256, 0, stream>>>(h2, mlp_w3, mlp_b3, mixb);

  // ---- bootstrap: mu, unbiased std -> d_out [1024 mu | 1024 std] ----
  k_musigma<<<blocks(1024), 256, 0, stream>>>(mixb, out);
}
